// W_MSA_21380347200235
// MI455X (gfx1250) — matrix operand order repaired and checked by reading, not compiled, not run on hardware
//
#include <hip/hip_runtime.h>
#include <math.h>
#include <stdint.h>

// ---------------------------------------------------------------------------
// CDNA5 / gfx1250 W-MSA block.
//   * bf16 WMMA GEMMs (v_wmma_f32_16x16x32_bf16), f32 accumulate
//   * Tensor Data Mover fills LDS tiles (tensor_load_to_lds, double-buffered,
//     BK=64) and drains bf16 output tiles (tensor_store_from_lds)
//   * all weights pre-transposed to [N][K] bf16 so A-tiles and B-tiles share
//     one descriptor shape (no LDS transpose, contiguous-K b128 fragments)
// wave32, 256-thread blocks (8 waves / block), 64 KB LDS of the 320 KB/WGP.
// ---------------------------------------------------------------------------

typedef __attribute__((ext_vector_type(16))) __bf16 v16bf;
typedef __attribute__((ext_vector_type(8)))  float  v8f;
typedef __attribute__((ext_vector_type(4)))  unsigned int u32x4;
typedef __attribute__((ext_vector_type(8)))  int          i32x8;
typedef __bf16 __attribute__((address_space(3))) lds_bf16_t;

union Frag { v16bf v; uint4 u[2]; };
union B8   { uint4 u; __bf16 e[8]; };

// LDS byte offset of a generic pointer into a __shared__ array.
static __device__ __forceinline__ unsigned lds_off(const __bf16* p)
{
    return (unsigned)(uintptr_t)(lds_bf16_t*)(__bf16*)p;
}

// ---- Tensor DMA descriptor (ISA 8.3 / 8.4), 2-D tile, bf16 elements -------
static __device__ __forceinline__ u32x4 tdm_g0(unsigned lds_addr,
                                               unsigned long long gaddr)
{
    u32x4 g;
    g.x = 1u;                                          // count=1, user mode
    g.y = lds_addr;                                    // lds_addr [63:32]
    g.z = (unsigned)(gaddr & 0xffffffffull);           // global_addr [95:64]
    g.w = (unsigned)((gaddr >> 32) & 0x01ffffffull)    // global_addr [120:96]
        | (2u << 30);                                  // type=2 ("image")
    return g;
}
static __device__ __forceinline__ i32x8 tdm_g1(unsigned dim0, unsigned dim1,
                                               unsigned tile0, unsigned tile1,
                                               unsigned stride0)
{
    i32x8 g;
    g[0] = (int)(1u << 16);                            // data_size=1 (2 bytes)
    g[1] = (int)((dim0 & 0xffffu) << 16);              // tensor_dim0 [79:48]
    g[2] = (int)((dim0 >> 16) | ((dim1 & 0xffffu) << 16)); // dim0 hi | dim1 lo
    g[3] = (int)((dim1 >> 16) | (tile0 << 16));        // dim1 hi | tile_dim0
    g[4] = (int)(tile1);                               // tile_dim1 (tile2=0)
    g[5] = (int)stride0;                               // tensor_dim0_stride lo
    g[6] = 0;
    g[7] = 0;
    return g;
}
static __device__ __forceinline__ void tdm_load(u32x4 g0, i32x8 g1)
{
    asm volatile("tensor_load_to_lds %0, %1" :: "s"(g0), "s"(g1) : "memory");
}
static __device__ __forceinline__ void tdm_store(u32x4 g0, i32x8 g1)
{
    asm volatile("tensor_store_from_lds %0, %1" :: "s"(g0), "s"(g1) : "memory");
}

// ---------------------------------------------------------------------------
// Tiled WMMA GEMM:  C[M,N] = A[M,K] * Bt[N,K]^T   (A, Bt bf16 row-major)
// Block tile 128x128, K-step 64, TDM double-buffered LDS staging.
// EPI: 0 = store bf16 via TDM              (QKV)
//      1 = +bias +resid -> f32             (proj, xr = xw + att@Wp + bp)
//      2 = +bias, exact GELU -> bf16 (TDM) (FFN mid)
//      3 = +bias +resid, window-unpartition scatter -> f32 out (final)
// ---------------------------------------------------------------------------
template<int EPI>
__global__ __launch_bounds__(256)
void gemm_bf16_wmma(const __bf16* __restrict__ A, const __bf16* __restrict__ Bt,
                    int M, int N, int K,
                    const float* __restrict__ bias,
                    const float* __restrict__ resid,
                    float* __restrict__ outF, __bf16* __restrict__ outB)
{
    // carved LDS: A tiles [2][128*64] at 0, B tiles [2][128*64] at 16384;
    // after the k-loop the first 16384 elems are reused as the bf16 out-tile.
    __shared__ __align__(16) __bf16 smem[32768];       // 64 KB

    const int tid  = threadIdx.x;
    const int lane = tid & 31;
    const int wid  = tid >> 5;
    const int wm   = wid >> 2;           // 0..1 : 64-row slab
    const int wn   = wid & 3;            // 0..3 : 32-col slab
    const int bm   = blockIdx.y;
    const int bn   = blockIdx.x;

    v8f acc[4][2];
#pragma unroll
    for (int i = 0; i < 4; ++i)
#pragma unroll
        for (int j = 0; j < 2; ++j)
#pragma unroll
            for (int e = 0; e < 8; ++e) acc[i][j][e] = 0.0f;

    const unsigned long long gaA =
        (unsigned long long)(uintptr_t)(A  + (size_t)bm * 128 * K);
    const unsigned long long gaB =
        (unsigned long long)(uintptr_t)(Bt + (size_t)bn * 128 * K);
    const unsigned ldsA[2] = { lds_off(&smem[0]),     lds_off(&smem[8192])  };
    const unsigned ldsB[2] = { lds_off(&smem[16384]), lds_off(&smem[24576]) };
    const i32x8 g1a = tdm_g1((unsigned)K, (unsigned)M, 64u, 128u, (unsigned)K);
    const i32x8 g1b = tdm_g1((unsigned)K, (unsigned)N, 64u, 128u, (unsigned)K);
    const int nk = K >> 6;               // 64-wide K stages

    if (wid == 0) {                      // prologue: stage 0 into buffer 0
        tdm_load(tdm_g0(ldsA[0], gaA), g1a);
        tdm_load(tdm_g0(ldsB[0], gaB), g1b);
    }

    const int kg    = lane >> 4;         // K-group within wave
    const int mlane = lane & 15;

    for (int it = 0; it < nk; ++it) {
        if (wid == 0) {
            if (it + 1 < nk) {           // prefetch stage it+1 into other buffer
                const unsigned long long koff = (unsigned long long)(it + 1) * 128;
                tdm_load(tdm_g0(ldsA[(it + 1) & 1], gaA + koff), g1a);
                tdm_load(tdm_g0(ldsB[(it + 1) & 1], gaB + koff), g1b);
                __builtin_amdgcn_s_wait_tensorcnt(2);   // stage 'it' complete
            } else {
                __builtin_amdgcn_s_wait_tensorcnt(0);
            }
        }
        __syncthreads();                 // LDS stage 'it' visible to all waves

        const __bf16* tA = &smem[(it & 1) * 8192];
        const __bf16* tB = &smem[16384 + (it & 1) * 8192];

#pragma unroll
        for (int kk = 0; kk < 2; ++kk) { // two 16x16x32 sub-steps per stage
            Frag afr[4], bfr[2];
            // A frag (ISA 7.12.2, 16-bit A 16x32): V0-3 K=kg*8.., V4-7 +16
#pragma unroll
            for (int tm = 0; tm < 4; ++tm) {
                const uint4* p = (const uint4*)
                    &tA[(wm * 64 + tm * 16 + mlane) * 64 + kk * 32 + kg * 8];
                afr[tm].u[0] = p[0];
                afr[tm].u[1] = p[2];
            }
            // B frag: lane holds column n, K = kg*8..+7 then +16 (same as A) in [n][k]
#pragma unroll
            for (int tn = 0; tn < 2; ++tn) {
                const uint4* p = (const uint4*)
                    &tB[(wn * 32 + tn * 16 + mlane) * 64 + kk * 32 + kg * 8];
                bfr[tn].u[0] = p[0];
                bfr[tn].u[1] = p[2];
            }
#pragma unroll
            for (int tm = 0; tm < 4; ++tm)
#pragma unroll
                for (int tn = 0; tn < 2; ++tn)
                    acc[tm][tn] = __builtin_amdgcn_wmma_f32_16x16x32_bf16(
                        false, afr[tm].v, false, bfr[tn].v,
                        (short)0, acc[tm][tn], false, false);
        }
        __syncthreads();                 // stage consumed; buffer reusable
    }

    // ---- epilogue (C/D layout: n = lane&15, m = v + 8*(lane>>4)) ----
    const int rh = lane >> 4;
    if (EPI == 0 || EPI == 2) {
        // stage bf16 tile in LDS (reuses A staging), drain with TDM store
        __bf16* outL = &smem[0];         // 128x128 bf16 = 32 KB
#pragma unroll
        for (int tm = 0; tm < 4; ++tm)
#pragma unroll
            for (int tn = 0; tn < 2; ++tn)
#pragma unroll
                for (int v = 0; v < 8; ++v) {
                    const int rl = wm * 64 + tm * 16 + v + 8 * rh;
                    const int cl = wn * 32 + tn * 16 + mlane;
                    float val = acc[tm][tn][v];
                    if (EPI == 2) {
                        val += bias[bn * 128 + cl];
                        val = 0.5f * val * (1.0f + erff(val * 0.70710678118654752f));
                    }
                    outL[rl * 128 + cl] = (__bf16)val;
                }
        __syncthreads();
        if (wid == 0) {
            const unsigned long long ga = (unsigned long long)(uintptr_t)
                (outB + (size_t)bm * 128 * N + bn * 128);
            tdm_store(tdm_g0(lds_off(outL), ga),
                      tdm_g1((unsigned)N, (unsigned)M, 128u, 128u, (unsigned)N));
            __builtin_amdgcn_s_wait_tensorcnt(0);
        }
    } else {
#pragma unroll
        for (int tm = 0; tm < 4; ++tm)
#pragma unroll
            for (int tn = 0; tn < 2; ++tn)
#pragma unroll
                for (int v = 0; v < 8; ++v) {
                    const int r = bm * 128 + wm * 64 + tm * 16 + v + 8 * rh;
                    const int c = bn * 128 + wn * 32 + tn * 16 + mlane;
                    float val = acc[tm][tn][v] + bias[c] + resid[(size_t)r * N + c];
                    if (EPI == 1) {
                        outF[(size_t)r * N + c] = val;
                    } else {  // EPI == 3: residual + reverse window partition
                        const int n_ = r >> 2, t_ = r & 3;
                        const int b_ = n_ >> 6, w_ = n_ & 63;
                        const int wr = w_ >> 3, wc = w_ & 7;
                        const int tok = ((wr * 2 + (t_ >> 1)) << 4) + (wc * 2 + (t_ & 1));
                        outF[((size_t)b_ * 256 + tok) * 1024 + c] = val;
                    }
                }
    }
}

// ---------------------------------------------------------------------------
// LayerNorm over C=1024 (one block per row, 256 threads x float4), optional
// window-partition row reorder, writes f32 and/or bf16 copies.
// ---------------------------------------------------------------------------
__global__ __launch_bounds__(256)
void ln_kernel(const float* __restrict__ x, const float* __restrict__ g,
               const float* __restrict__ b, float* __restrict__ outF,
               __bf16* __restrict__ outB, int window_reorder)
{
    __shared__ float ss[256], sq[256];
    const int tid = threadIdx.x;
    const int row = blockIdx.x;
    float4 xv = ((const float4*)(x + (size_t)row * 1024))[tid];
    ss[tid] = xv.x + xv.y + xv.z + xv.w;
    sq[tid] = xv.x * xv.x + xv.y * xv.y + xv.z * xv.z + xv.w * xv.w;
    __syncthreads();
    for (int st = 128; st > 0; st >>= 1) {
        if (tid < st) { ss[tid] += ss[tid + st]; sq[tid] += sq[tid + st]; }
        __syncthreads();
    }
    const float mean = ss[0] * (1.0f / 1024.0f);
    const float var  = sq[0] * (1.0f / 1024.0f) - mean * mean;
    const float rstd = rsqrtf(var + 1e-5f);

    const float4 gv = ((const float4*)g)[tid];
    const float4 bv = ((const float4*)b)[tid];
    float4 y;
    y.x = (xv.x - mean) * rstd * gv.x + bv.x;
    y.y = (xv.y - mean) * rstd * gv.y + bv.y;
    y.z = (xv.z - mean) * rstd * gv.z + bv.z;
    y.w = (xv.w - mean) * rstd * gv.w + bv.w;

    int r = row;
    if (window_reorder) {                       // token order -> window order
        const int b_ = row >> 8, t = row & 255;
        const int pr = t >> 4, pc = t & 15;
        r = ((b_ * 64 + (pr >> 1) * 8 + (pc >> 1)) << 2) + ((pr & 1) * 2 + (pc & 1));
    }
    if (outF) ((float4*)(outF + (size_t)r * 1024))[tid] = y;
    B8 pk;
    pk.e[0] = (__bf16)y.x; pk.e[1] = (__bf16)y.y;
    pk.e[2] = (__bf16)y.z; pk.e[3] = (__bf16)y.w;
    ((uint2*)(outB + (size_t)r * 1024))[tid] = *(uint2*)&pk.u;
}

// ---------------------------------------------------------------------------
// 4x4 window attention: one thread per (window, head, query).
// qkv is [16384, 3072] bf16 (cols: 0..1023 Q | 1024..2047 K | 2048..3071 V),
// head-major within each third.  scale = C^-0.5 = 1/32 (faithful quirk).
// ---------------------------------------------------------------------------
__global__ __launch_bounds__(256)
void attn_kernel(const __bf16* __restrict__ qkv, __bf16* __restrict__ att)
{
    const int idx = blockIdx.x * 256 + threadIdx.x;     // 4096*16*4 threads
    const int qi = idx & 3, hh = (idx >> 2) & 15, n = idx >> 6;

    const uint4* qp = (const uint4*)(qkv + (size_t)(n * 4 + qi) * 3072 + hh * 64);
    float qr[64];
#pragma unroll
    for (int c = 0; c < 8; ++c) {
        B8 t; t.u = qp[c];
#pragma unroll
        for (int e = 0; e < 8; ++e) qr[c * 8 + e] = (float)t.e[e];
    }
    float s[4];
#pragma unroll
    for (int j = 0; j < 4; ++j) {
        const uint4* kp = (const uint4*)(qkv + (size_t)(n * 4 + j) * 3072 + 1024 + hh * 64);
        float a = 0.0f;
#pragma unroll
        for (int c = 0; c < 8; ++c) {
            B8 t; t.u = kp[c];
#pragma unroll
            for (int e = 0; e < 8; ++e) a += qr[c * 8 + e] * (float)t.e[e];
        }
        s[j] = a * 0.03125f;
    }
    const float mx = fmaxf(fmaxf(s[0], s[1]), fmaxf(s[2], s[3]));
    float w[4], den = 0.0f;
#pragma unroll
    for (int j = 0; j < 4; ++j) { w[j] = __expf(s[j] - mx); den += w[j]; }
    const float inv = 1.0f / den;

    float o[64];
#pragma unroll
    for (int d = 0; d < 64; ++d) o[d] = 0.0f;
#pragma unroll
    for (int j = 0; j < 4; ++j) {
        const uint4* vp = (const uint4*)(qkv + (size_t)(n * 4 + j) * 3072 + 2048 + hh * 64);
        const float wj = w[j] * inv;
#pragma unroll
        for (int c = 0; c < 8; ++c) {
            B8 t; t.u = vp[c];
#pragma unroll
            for (int e = 0; e < 8; ++e) o[c * 8 + e] += wj * (float)t.e[e];
        }
    }
    uint4* op = (uint4*)(att + (size_t)(n * 4 + qi) * 1024 + hh * 64);
#pragma unroll
    for (int c = 0; c < 8; ++c) {
        B8 t;
#pragma unroll
        for (int e = 0; e < 8; ++e) t.e[e] = (__bf16)o[c * 8 + e];
        op[c] = t.u;
    }
}

// ---------------------------------------------------------------------------
// Weight prep: all GEMM B-operands stored TRANSPOSED bf16 [N][K] so GEMM
// A-tiles and B-tiles share one TDM descriptor shape.
// ---------------------------------------------------------------------------
__global__ __launch_bounds__(256)
void prep_qkv_wT(const float* __restrict__ Wq, const float* __restrict__ Wk,
                 const float* __restrict__ Wv, __bf16* __restrict__ dst)
{
    const int idx = blockIdx.x * 256 + threadIdx.x;     // [3072][1024]
    if (idx >= 3072 * 1024) return;
    const int col = idx >> 10, c = idx & 1023;          // col = output feature
    const int sel = col >> 10, cc = col & 1023;
    const int h = cc >> 6, d = cc & 63;
    const float* W = (sel == 0) ? Wq : (sel == 1) ? Wk : Wv;
    dst[idx] = (__bf16)W[((size_t)h * 1024 + c) * 64 + d];
}

__global__ __launch_bounds__(256)
void cvt_bf16_T(const float* __restrict__ src, __bf16* __restrict__ dst)
{
    const int idx = blockIdx.x * 256 + threadIdx.x;     // 1024x1024 transpose
    const int n = idx >> 10, k = idx & 1023;
    dst[idx] = (__bf16)src[(size_t)k * 1024 + n];
}

// ---------------------------------------------------------------------------
extern "C" void kernel_launch(void* const* d_in, const int* in_sizes, int n_in,
                              void* d_out, int out_size, void* d_ws, size_t ws_size,
                              hipStream_t stream)
{
    const float* x     = (const float*)d_in[0];
    const float* ln1_g = (const float*)d_in[1];
    const float* ln1_b = (const float*)d_in[2];
    const float* Wq    = (const float*)d_in[3];
    const float* Wk    = (const float*)d_in[4];
    const float* Wv    = (const float*)d_in[5];
    const float* Wp    = (const float*)d_in[6];
    const float* bp    = (const float*)d_in[7];
    const float* ln2_g = (const float*)d_in[8];
    const float* ln2_b = (const float*)d_in[9];
    const float* W1    = (const float*)d_in[10];
    const float* b1    = (const float*)d_in[11];
    const float* W2    = (const float*)d_in[12];
    const float* b2    = (const float*)d_in[13];

    constexpr size_t R = 16384;       // B * T rows
    char* ws = (char*)d_ws;
    size_t off = 0;
    auto alloc = [&](size_t bytes) -> void* {
        off = (off + 255) & ~(size_t)255;
        void* p = ws + off;
        off += bytes;
        return p;
    };
    float*  xw_f32 = (float*) alloc(R * 1024 * 4);      // LN'd, window-ordered
    __bf16* xw_bf  = (__bf16*)alloc(R * 1024 * 2);
    __bf16* WqkvT  = (__bf16*)alloc(3072ull * 1024 * 2);
    __bf16* WpT    = (__bf16*)alloc(1024ull * 1024 * 2);
    __bf16* W1T    = (__bf16*)alloc(1024ull * 1024 * 2);
    __bf16* W2T    = (__bf16*)alloc(1024ull * 1024 * 2);
    __bf16* qkv    = (__bf16*)alloc(R * 3072 * 2);
    __bf16* att    = (__bf16*)alloc(R * 1024 * 2);
    float*  xr     = (float*) alloc(R * 1024 * 4);      // window-ordered resid
    __bf16* h2     = (__bf16*)alloc(R * 1024 * 2);      // LN2 output
    __bf16* g2     = (__bf16*)alloc(R * 1024 * 2);      // GELU(FFN1) output

    // 1) weight prep (transposed bf16, 12 MB total -> L2-resident)
    prep_qkv_wT<<<(3072 * 1024 + 255) / 256, 256, 0, stream>>>(Wq, Wk, Wv, WqkvT);
    cvt_bf16_T<<<(1024 * 1024 + 255) / 256, 256, 0, stream>>>(Wp, WpT);
    cvt_bf16_T<<<(1024 * 1024 + 255) / 256, 256, 0, stream>>>(W1, W1T);
    cvt_bf16_T<<<(1024 * 1024 + 255) / 256, 256, 0, stream>>>(W2, W2T);

    // 2) LN1 + window partition (+ f32 copy for the faithful residual)
    ln_kernel<<<R, 256, 0, stream>>>(x, ln1_g, ln1_b, xw_f32, xw_bf, 1);

    // 3) fused QKV GEMM: [16384,1024] x [1024,3072] -> bf16 (TDM out)
    gemm_bf16_wmma<0><<<dim3(3072 / 128, R / 128), 256, 0, stream>>>(
        xw_bf, WqkvT, (int)R, 3072, 1024, nullptr, nullptr, nullptr, qkv);

    // 4) 4x4 windowed attention (tiny; VALU)
    attn_kernel<<<(R * 16) / 256, 256, 0, stream>>>(qkv, att);

    // 5) proj GEMM + bias + residual (xr = xw + att@Wp + bp), f32
    gemm_bf16_wmma<1><<<dim3(1024 / 128, R / 128), 256, 0, stream>>>(
        att, WpT, (int)R, 1024, 1024, bp, xw_f32, xr, nullptr);

    // 6) LN2 (window order irrelevant for per-token ops)
    ln_kernel<<<R, 256, 0, stream>>>(xr, ln2_g, ln2_b, nullptr, h2, 0);

    // 7) FFN1 + exact GELU -> bf16 (TDM out)
    gemm_bf16_wmma<2><<<dim3(1024 / 128, R / 128), 256, 0, stream>>>(
        h2, W1T, (int)R, 1024, 1024, b1, nullptr, nullptr, g2);

    // 8) FFN2 + bias + residual + reverse window partition -> d_out (f32)
    gemm_bf16_wmma<3><<<dim3(1024 / 128, R / 128), 256, 0, stream>>>(
        g2, W2T, (int)R, 1024, 1024, b2, xr, (float*)d_out, nullptr);
}
